// EGNN_21045339750997
// MI455X (gfx1250) — matrix operand order repaired and checked by reading, not compiled, not run on hardware
//
#include <hip/hip_runtime.h>
#include <hip/hip_bf16.h>

typedef __attribute__((ext_vector_type(16))) __bf16 v16bf;
typedef __attribute__((ext_vector_type(8)))  float  v8f;

#define WAVES 4
#define LDW   68   // padded LDS row stride (floats): (4*row+col)%64 -> conflict-free

struct EdgeLds {
    float cd [16][12];    // per-edge coord diffs (K*3 = 12)
    float A2 [16][LDW];   // SiLU(layer1) activations
    float MIJ[16][LDW];   // edge messages mij (f32)
    float H  [16][LDW];   // hidden buffer (gate / px, reused)
    float gij[16];
    float wc [16][4];     // px coord weights (also reused as dot partials)
};

__device__ __forceinline__ float silu_f(float v) { return v / (1.0f + __expf(-v)); }

// A-fragment (16x32, bf16) built from a row of f32 values (global or LDS).
// Lane l: row = l&15; elems[0..7] = K kb..kb+7, elems[8..15] = K kb+16..kb+23,
// kb = kc*32 + (l>=16 ? 8 : 0)   (per CDNA5 ISA 16-bit A layout)
__device__ __forceinline__ v16bf frag_a_row(const float* row, int kc, int hl) {
    const int kb = kc * 32 + hl * 8;
    v16bf a;
#pragma unroll
    for (int i = 0; i < 8; ++i) {
        a[i]     = (__bf16)row[kb + i];
        a[8 + i] = (__bf16)row[kb + 16 + i];
    }
    return a;
}

// B-fragment (32x16, bf16) from transposed weights Bt[64][Kdim].
// Lane l: col n = nt*16 + (l&15); elems[0..7] = K kb..kb+7, elems[8..15] = K kb+16..kb+23, kb = kc*32 + (l>=16?8:0)
__device__ __forceinline__ v16bf frag_b(const __bf16* __restrict__ Bt, int Kdim,
                                        int nt, int kc, int lane) {
    const int n  = nt * 16 + (lane & 15);
    const int kb = kc * 32 + ((lane >> 4) & 1) * 8;
    const __bf16* p = Bt + n * Kdim + kb;
    v16bf b;
#pragma unroll
    for (int i = 0; i < 8; ++i) { b[i] = p[i]; b[8 + i] = p[16 + i]; }
    return b;
}

// ---------------------------------------------------------------------------
// Edge kernel: 16 edges per wave, 4 waves per block.
// ---------------------------------------------------------------------------
__global__ __launch_bounds__(128) void egnn_edge(
    const float* __restrict__ h, const float* __restrict__ x,
    const float* __restrict__ edge_attr, const int* __restrict__ ei,
    const float* __restrict__ pm_b1, const float* __restrict__ pm_b2,
    const float* __restrict__ mg_b1, const float* __restrict__ mg_w2,
    const float* __restrict__ mg_b2,
    const float* __restrict__ px_b1, const float* __restrict__ px_w2,
    const float* __restrict__ px_b2,
    const __bf16* __restrict__ pm_w1t, const __bf16* __restrict__ pm_w2t,
    const __bf16* __restrict__ mg_w1t, const __bf16* __restrict__ px_w1t,
    float* __restrict__ h_agg, float* __restrict__ x_agg, float* __restrict__ cnt,
    int E)
{
    __shared__ EdgeLds lds[WAVES];
    const int lane = threadIdx.x & 31;
    const int wave = threadIdx.x >> 5;
    const int m    = lane & 15;     // edge row within tile
    const int hl   = lane >> 4;     // K-half selector
    EdgeLds& L = lds[wave];

    const long tile  = (long)blockIdx.x * WAVES + wave;
    const long e     = tile * 16 + m;
    const bool valid = (e < (long)E);
    const long ec    = valid ? e : (long)(E - 1);
    const int  src   = ei[ec];
    const int  dst   = ei[(long)E + ec];

    // coord_diff -> LDS (one lane per edge)
    if (hl == 0) {
#pragma unroll
        for (int j = 0; j < 12; ++j)
            L.cd[m][j] = x[(long)src * 12 + j] - x[(long)dst * 12 + j];
    }
    __syncthreads();

    // radial Gram entries: this lane's 8 of 16 (flattened i*4+j)
    float r8[8];
    {
        float cdl[12];
#pragma unroll
        for (int j = 0; j < 12; ++j) cdl[j] = L.cd[m][j];
#pragma unroll
        for (int t = 0; t < 8; ++t) {
            const int g = hl * 8 + t, gi = g >> 2, gj = g & 3;
            r8[t] = cdl[gi*3+0]*cdl[gj*3+0] + cdl[gi*3+1]*cdl[gj*3+1]
                  + cdl[gi*3+2]*cdl[gj*3+2];
        }
    }

    // ---- Layer 1: [h_src | h_dst | radial | edge_attr] (K=160) @ pm_w1 -> SiLU
    v16bf aCh[5];
    {
        const float* hs = h + (long)src * 64;
        const float* hd = h + (long)dst * 64;
        aCh[0] = frag_a_row(hs, 0, hl);
        aCh[1] = frag_a_row(hs, 1, hl);
        aCh[2] = frag_a_row(hd, 0, hl);
        aCh[3] = frag_a_row(hd, 1, hl);
        v16bf a4;
#pragma unroll
        for (int i = 0; i < 8; ++i) {
            a4[i]     = (__bf16)r8[i];                           // K 128..143
            a4[8 + i] = (__bf16)edge_attr[ec * 16 + hl * 8 + i]; // K 144..159
        }
        aCh[4] = a4;
    }
#pragma unroll
    for (int nt = 0; nt < 4; ++nt) {
        v8f c = {};
#pragma unroll
        for (int kc = 0; kc < 5; ++kc) {
            v16bf b = frag_b(pm_w1t, 160, nt, kc, lane);
            c = __builtin_amdgcn_wmma_f32_16x16x32_bf16(false, aCh[kc], false, b,
                                                        (short)0, c, false, false);
        }
        const float bias = pm_b1[nt * 16 + m];
#pragma unroll
        for (int i = 0; i < 8; ++i)       // D: row = hl*8+i, col = nt*16+m
            L.A2[hl * 8 + i][nt * 16 + m] = silu_f(c[i] + bias);
    }
    __syncthreads();

    // ---- Layer 2: mij = A2 @ pm_w2 + b2
#pragma unroll
    for (int nt = 0; nt < 4; ++nt) {
        v8f c = {};
#pragma unroll
        for (int kc = 0; kc < 2; ++kc) {
            v16bf a = frag_a_row(&L.A2[m][0], kc, hl);
            v16bf b = frag_b(pm_w2t, 64, nt, kc, lane);
            c = __builtin_amdgcn_wmma_f32_16x16x32_bf16(false, a, false, b,
                                                        (short)0, c, false, false);
        }
        const float bias = pm_b2[nt * 16 + m];
#pragma unroll
        for (int i = 0; i < 8; ++i)
            L.MIJ[hl * 8 + i][nt * 16 + m] = c[i] + bias;
    }
    __syncthreads();

    // ---- Gate MLP hidden: SiLU(mij @ mg_w1 + b1)
#pragma unroll
    for (int nt = 0; nt < 4; ++nt) {
        v8f c = {};
#pragma unroll
        for (int kc = 0; kc < 2; ++kc) {
            v16bf a = frag_a_row(&L.MIJ[m][0], kc, hl);
            v16bf b = frag_b(mg_w1t, 64, nt, kc, lane);
            c = __builtin_amdgcn_wmma_f32_16x16x32_bf16(false, a, false, b,
                                                        (short)0, c, false, false);
        }
        const float bias = mg_b1[nt * 16 + m];
#pragma unroll
        for (int i = 0; i < 8; ++i)
            L.H[hl * 8 + i][nt * 16 + m] = silu_f(c[i] + bias);
    }
    __syncthreads();
    // gij = H @ mg_w2 + b2  (64 -> 1; split dot over the two half-lanes)
    {
        float part = 0.f;
#pragma unroll
        for (int k2 = 0; k2 < 32; ++k2) {
            const int k = hl * 32 + k2;
            part += L.H[m][k] * mg_w2[k];
        }
        L.wc[m][hl] = part;
    }
    __syncthreads();
    if (lane < 16) L.gij[m] = L.wc[m][0] + L.wc[m][1] + mg_b2[0];
    __syncthreads();

    // ---- Coord MLP hidden: SiLU(mij @ px_w1 + b1)
#pragma unroll
    for (int nt = 0; nt < 4; ++nt) {
        v8f c = {};
#pragma unroll
        for (int kc = 0; kc < 2; ++kc) {
            v16bf a = frag_a_row(&L.MIJ[m][0], kc, hl);
            v16bf b = frag_b(px_w1t, 64, nt, kc, lane);
            c = __builtin_amdgcn_wmma_f32_16x16x32_bf16(false, a, false, b,
                                                        (short)0, c, false, false);
        }
        const float bias = px_b1[nt * 16 + m];
#pragma unroll
        for (int i = 0; i < 8; ++i)
            L.H[hl * 8 + i][nt * 16 + m] = silu_f(c[i] + bias);
    }
    __syncthreads();
    // w = H @ px_w2 + b2  (64 -> 4; 2 coefs per lane)
#pragma unroll
    for (int cc = 0; cc < 2; ++cc) {
        const int ch = hl * 2 + cc;
        float s = px_b2[ch];
#pragma unroll 8
        for (int k = 0; k < 64; ++k) s += L.H[m][k] * px_w2[k * 4 + ch];
        L.wc[m][ch] = s;
    }
    __syncthreads();

    // ---- Scatters (atomic f32)
    if (valid) {
        const float g  = L.gij[m];
        const long  hb = (long)src * 64;
#pragma unroll
        for (int i2 = 0; i2 < 32; ++i2) {
            const int col = hl * 32 + i2;
            atomicAdd(&h_agg[hb + col], L.MIJ[m][col] * g);
        }
        if (hl == 0) atomicAdd(&cnt[src], 1.0f);
#pragma unroll
        for (int j = 0; j < 6; ++j) {
            const int jj = hl * 6 + j;
            atomicAdd(&x_agg[(long)src * 12 + jj], L.cd[m][jj] * L.wc[m][jj / 3]);
        }
    }
}

// ---------------------------------------------------------------------------
// Node kernel: h_new = h + MLP([h_agg | h | x(12) | pad], K padded to 160)
// ---------------------------------------------------------------------------
__global__ __launch_bounds__(128) void egnn_node(
    const float* __restrict__ h, const float* __restrict__ x,
    const float* __restrict__ h_agg,
    const float* __restrict__ ph_b1, const float* __restrict__ ph_b2,
    const __bf16* __restrict__ ph_w1t, const __bf16* __restrict__ ph_w2t,
    float* __restrict__ hnew, int N)
{
    __shared__ float Hld[WAVES][16][LDW];
    const int lane = threadIdx.x & 31;
    const int wave = threadIdx.x >> 5;
    const int m    = lane & 15;
    const int hl   = lane >> 4;

    const long tile = (long)blockIdx.x * WAVES + wave;
    const long n0   = tile * 16;
    const long node = n0 + m;
    const long nc   = (node < (long)N) ? node : (long)(N - 1);

    v16bf aCh[5];
    {
        const float* ha = h_agg + nc * 64;
        const float* hh = h     + nc * 64;
        const float* xx = x     + nc * 12;
        aCh[0] = frag_a_row(ha, 0, hl);
        aCh[1] = frag_a_row(ha, 1, hl);
        aCh[2] = frag_a_row(hh, 0, hl);
        aCh[3] = frag_a_row(hh, 1, hl);
        v16bf a4;
#pragma unroll
        for (int i = 0; i < 8; ++i) {
            const int idx = hl * 8 + i;                    // K 128..143 -> x flat
            a4[i]     = (idx < 12) ? (__bf16)xx[idx] : (__bf16)0.0f;
            a4[8 + i] = (__bf16)0.0f;                      // K 144..159 -> pad
        }
        aCh[4] = a4;
    }
#pragma unroll
    for (int nt = 0; nt < 4; ++nt) {
        v8f c = {};
#pragma unroll
        for (int kc = 0; kc < 5; ++kc) {
            v16bf b = frag_b(ph_w1t, 160, nt, kc, lane);
            c = __builtin_amdgcn_wmma_f32_16x16x32_bf16(false, aCh[kc], false, b,
                                                        (short)0, c, false, false);
        }
        const float bias = ph_b1[nt * 16 + m];
#pragma unroll
        for (int i = 0; i < 8; ++i)
            Hld[wave][hl * 8 + i][nt * 16 + m] = silu_f(c[i] + bias);
    }
    __syncthreads();
#pragma unroll
    for (int nt = 0; nt < 4; ++nt) {
        v8f c = {};
#pragma unroll
        for (int kc = 0; kc < 2; ++kc) {
            v16bf a = frag_a_row(&Hld[wave][m][0], kc, hl);
            v16bf b = frag_b(ph_w2t, 64, nt, kc, lane);
            c = __builtin_amdgcn_wmma_f32_16x16x32_bf16(false, a, false, b,
                                                        (short)0, c, false, false);
        }
        const float bias = ph_b2[nt * 16 + m];
        const int   col  = nt * 16 + m;
#pragma unroll
        for (int i = 0; i < 8; ++i) {
            const long nr = n0 + hl * 8 + i;               // D row -> node
            if (nr < (long)N)
                hnew[nr * 64 + col] = h[nr * 64 + col] + c[i] + bias;
        }
    }
}

// ---------------------------------------------------------------------------
// x_new = x + clip(x_agg / max(cnt,1), -10, 10)
// ---------------------------------------------------------------------------
__global__ void egnn_xnew(const float* __restrict__ x, const float* __restrict__ x_agg,
                          const float* __restrict__ cnt, float* __restrict__ out, long total)
{
    for (long i = (long)blockIdx.x * blockDim.x + threadIdx.x; i < total;
         i += (long)gridDim.x * blockDim.x) {
        const long node = i / 12;
        float c = cnt[node];
        c = (c < 1.0f) ? 1.0f : c;
        float v = x_agg[i] / c;
        v = fminf(fmaxf(v, -10.0f), 10.0f);
        out[i] = x[i] + v;
    }
}

__global__ void egnn_zero(float* __restrict__ p, long total)
{
    for (long i = (long)blockIdx.x * blockDim.x + threadIdx.x; i < total;
         i += (long)gridDim.x * blockDim.x)
        p[i] = 0.0f;
}

// Weight prep: transpose + convert to bf16 (ph_w1 also permuted/padded to K=160).
__global__ void egnn_prep(const float* __restrict__ pm_w1, const float* __restrict__ pm_w2,
                          const float* __restrict__ mg_w1, const float* __restrict__ px_w1,
                          const float* __restrict__ ph_w1, const float* __restrict__ ph_w2,
                          __bf16* __restrict__ pm_w1t, __bf16* __restrict__ pm_w2t,
                          __bf16* __restrict__ mg_w1t, __bf16* __restrict__ px_w1t,
                          __bf16* __restrict__ ph_w1t, __bf16* __restrict__ ph_w2t)
{
    const int tid    = blockIdx.x * blockDim.x + threadIdx.x;
    const int stride = gridDim.x * blockDim.x;
    for (int idx = tid; idx < 64 * 160; idx += stride) {
        const int n = idx / 160, k = idx % 160;
        pm_w1t[idx] = (__bf16)pm_w1[k * 64 + n];
        // A order: [h_agg(0..63) | h(64..127) | x(128..139) | pad] -> orig rows
        float v = 0.0f;
        if (k < 128)      v = ph_w1[(12 + k) * 64 + n];
        else if (k < 140) v = ph_w1[(k - 128) * 64 + n];
        ph_w1t[idx] = (__bf16)v;
    }
    for (int idx = tid; idx < 64 * 64; idx += stride) {
        const int n = idx / 64, k = idx % 64;
        pm_w2t[idx] = (__bf16)pm_w2[k * 64 + n];
        mg_w1t[idx] = (__bf16)mg_w1[k * 64 + n];
        px_w1t[idx] = (__bf16)px_w1[k * 64 + n];
        ph_w2t[idx] = (__bf16)ph_w2[k * 64 + n];
    }
}

extern "C" void kernel_launch(void* const* d_in, const int* in_sizes, int n_in,
                              void* d_out, int out_size, void* d_ws, size_t ws_size,
                              hipStream_t stream) {
    const float* h         = (const float*)d_in[0];
    const float* x         = (const float*)d_in[1];
    const float* edge_attr = (const float*)d_in[2];
    // d_in[3] = mask (unused by the reference math)
    const float* pm_w1 = (const float*)d_in[4];
    const float* pm_b1 = (const float*)d_in[5];
    const float* pm_w2 = (const float*)d_in[6];
    const float* pm_b2 = (const float*)d_in[7];
    const float* mg_w1 = (const float*)d_in[8];
    const float* mg_b1 = (const float*)d_in[9];
    const float* mg_w2 = (const float*)d_in[10];
    const float* mg_b2 = (const float*)d_in[11];
    const float* px_w1 = (const float*)d_in[12];
    const float* px_b1 = (const float*)d_in[13];
    const float* px_w2 = (const float*)d_in[14];
    const float* px_b2 = (const float*)d_in[15];
    const float* ph_w1 = (const float*)d_in[16];
    const float* ph_b1 = (const float*)d_in[17];
    const float* ph_w2 = (const float*)d_in[18];
    const float* ph_b2 = (const float*)d_in[19];
    const int*   ei    = (const int*)d_in[20];

    const int N = in_sizes[0] / 64;
    const int E = in_sizes[20] / 2;

    // Workspace layout: [h_agg N*64 | x_agg N*12 | cnt N] f32, then bf16 weights
    float* ws     = (float*)d_ws;
    float* h_agg  = ws;
    float* x_agg  = ws + (long)N * 64;
    float* cntp   = ws + (long)N * 76;
    const size_t zfloats = (size_t)N * 77;
    size_t wbytes = (zfloats * 4 + 255) & ~(size_t)255;
    __bf16* pm_w1t = (__bf16*)((char*)d_ws + wbytes);
    __bf16* pm_w2t = pm_w1t + 64 * 160;
    __bf16* mg_w1t = pm_w2t + 64 * 64;
    __bf16* px_w1t = mg_w1t + 64 * 64;
    __bf16* ph_w1t = px_w1t + 64 * 64;
    __bf16* ph_w2t = ph_w1t + 64 * 160;

    {
        long total = (long)zfloats;
        int blocks = (int)((total + 255) / 256);
        if (blocks > 4096) blocks = 4096;
        egnn_zero<<<blocks, 256, 0, stream>>>(ws, total);
    }
    egnn_prep<<<64, 256, 0, stream>>>(pm_w1, pm_w2, mg_w1, px_w1, ph_w1, ph_w2,
                                      pm_w1t, pm_w2t, mg_w1t, px_w1t, ph_w1t, ph_w2t);

    {
        long tiles  = ((long)E + 15) / 16;
        int  blocks = (int)((tiles + WAVES - 1) / WAVES);
        egnn_edge<<<blocks, 32 * WAVES, 0, stream>>>(
            h, x, edge_attr, ei,
            pm_b1, pm_b2, mg_b1, mg_w2, mg_b2, px_b1, px_w2, px_b2,
            pm_w1t, pm_w2t, mg_w1t, px_w1t,
            h_agg, x_agg, cntp, E);
    }
    {
        long tiles  = ((long)N + 15) / 16;
        int  blocks = (int)((tiles + WAVES - 1) / WAVES);
        egnn_node<<<blocks, 32 * WAVES, 0, stream>>>(
            h, x, h_agg, ph_b1, ph_b2, ph_w1t, ph_w2t, (float*)d_out, N);
    }
    {
        long total = (long)N * 12;
        int blocks = (int)((total + 255) / 256);
        if (blocks > 4096) blocks = 4096;
        egnn_xnew<<<blocks, 256, 0, stream>>>(x, x_agg, cntp,
                                              (float*)d_out + (long)N * 64, total);
    }
}
